// Mymodel_13632226198173
// MI455X (gfx1250) — matrix operand order repaired and checked by reading, not compiled, not run on hardware
//
#include <hip/hip_runtime.h>
#include <hip/hip_bf16.h>
#include <math.h>
#include <stdint.h>

typedef __attribute__((ext_vector_type(16))) _Float16 v16h;
typedef __attribute__((ext_vector_type(8)))  _Float16 v8h;
typedef __attribute__((ext_vector_type(8)))  float    v8f;

#define B_    1024
#define L_    50
#define EMB_  63
#define CD_   60
#define H_    100
#define D_    200
#define KSEQ  256    // 243 padded to mult of 32
#define NGATE 800    // 4H * 2 directions
#define KHID  128    // 100 padded
#define KH2   224    // 200 padded
#define NQKV  608    // 600 padded to mult of 16
#define KATT  10016  // 10000 padded
#define BROW  40     // LDS B-strip row stride in halves (80B: 16B-aligned, bank-spread)

union Frag { v16h v; v8h h[2]; };

__device__ __forceinline__ float sigf(float x)     { return 1.f / (1.f + __expf(-x)); }
__device__ __forceinline__ float tanhfast(float x) { return 2.f / (1.f + __expf(-2.f * x)) - 1.f; }

// ---------------- embeddings -> seq16[b][t][0..62] (f16, row stride KSEQ) ------------
__global__ void embed_kernel(const int* __restrict__ x, const float* __restrict__ emb1,
                             const float* __restrict__ emb2, _Float16* __restrict__ seq16) {
  int idx = blockIdx.x * blockDim.x + threadIdx.x;   // b*L + t
  if (idx >= B_ * L_) return;
  int tok = x[idx];
  _Float16* row = seq16 + (size_t)idx * KSEQ;
  #pragma unroll
  for (int c = 0; c < 21; c++) row[c] = (_Float16)emb1[tok * 21 + c];
  #pragma unroll
  for (int c = 0; c < 42; c++) row[21 + c] = (_Float16)emb2[tok * 42 + c];
}

// ---------------- conv1d VALID + ReLU (raw, pre-pool) --------------------------------
__global__ void conv_kernel(const _Float16* __restrict__ seq16, const float* __restrict__ w,
                            const float* __restrict__ bias, float* __restrict__ out,
                            int h, int Lout) {
  int idx = blockIdx.x * blockDim.x + threadIdx.x;
  int total = B_ * CD_ * Lout;
  if (idx >= total) return;
  int t0 = idx % Lout;
  int co = (idx / Lout) % CD_;
  int b  = idx / (Lout * CD_);
  float acc = bias[co];
  const float* wc = w + (size_t)co * EMB_ * h;
  const _Float16* srow = seq16 + ((size_t)b * L_ + t0) * KSEQ;
  for (int tap = 0; tap < h; tap++) {
    const _Float16* s = srow + tap * KSEQ;
    for (int c = 0; c < EMB_; c++)
      acc += (float)s[c] * wc[c * h + tap];
  }
  out[idx] = fmaxf(acc, 0.f);
}

// ---------------- maxpool(k=5,pad=1,stride=1) + write padded channels into seq16 -----
__global__ void pool_kernel(const float* __restrict__ conv, _Float16* __restrict__ seq16,
                            int Lc, int chanOff) {
  int Lo = Lc - 2;
  int idx = blockIdx.x * blockDim.x + threadIdx.x;
  int total = B_ * CD_ * Lo;
  if (idx >= total) return;
  int t  = idx % Lo;
  int co = (idx / Lo) % CD_;
  int b  = idx / (Lo * CD_);
  const float* base = conv + ((size_t)b * CD_ + co) * Lc;
  int lo = t - 1 < 0 ? 0 : t - 1;
  int hi = t + 3 > Lc - 1 ? Lc - 1 : t + 3;
  float m = base[lo];
  for (int j = lo + 1; j <= hi; j++) m = fmaxf(m, base[j]);
  seq16[((size_t)b * L_ + t) * KSEQ + chanOff + co] = (_Float16)m;
}

// ---------------- weight conversion / padding to f16 ---------------------------------
__global__ void prep_kernel(const float* __restrict__ wih_f, const float* __restrict__ wih_b,
                            const float* __restrict__ bih_f, const float* __restrict__ bhh_f,
                            const float* __restrict__ bih_b, const float* __restrict__ bhh_b,
                            const float* __restrict__ whh_f, const float* __restrict__ whh_b,
                            const float* __restrict__ wq, const float* __restrict__ wk,
                            const float* __restrict__ wv, const float* __restrict__ bq,
                            const float* __restrict__ bk, const float* __restrict__ bv,
                            const float* __restrict__ dw1,
                            _Float16* WihT, float* gxbias, _Float16* WhhF, _Float16* WhhB,
                            _Float16* Wqkv, float* bqkv, _Float16* dw1t) {
  long long i = (long long)blockIdx.x * blockDim.x + threadIdx.x;
  const long long S0 = 800LL * KSEQ;
  const long long S1 = S0 + 800;
  const long long S2 = S1 + 400LL * KHID;
  const long long S3 = S2 + 400LL * KHID;
  const long long S4 = S3 + (long long)NQKV * KH2;
  const long long S5 = S4 + NQKV;
  const long long S6 = S5 + 128LL * KATT;
  if (i < S0) {
    int n = (int)(i / KSEQ), k = (int)(i % KSEQ);
    float v = 0.f;
    if (k < 243) v = (n < 400) ? wih_f[n * 243 + k] : wih_b[(n - 400) * 243 + k];
    WihT[i] = (_Float16)v;
  } else if (i < S1) {
    int n = (int)(i - S0);
    gxbias[n] = (n < 400) ? (bih_f[n] + bhh_f[n]) : (bih_b[n - 400] + bhh_b[n - 400]);
  } else if (i < S2) {
    long long r = i - S1; int n = (int)(r / KHID), k = (int)(r % KHID);
    WhhF[r] = (_Float16)((k < 100) ? whh_f[n * 100 + k] : 0.f);
  } else if (i < S3) {
    long long r = i - S2; int n = (int)(r / KHID), k = (int)(r % KHID);
    WhhB[r] = (_Float16)((k < 100) ? whh_b[n * 100 + k] : 0.f);
  } else if (i < S4) {
    long long r = i - S3; int n = (int)(r / KH2), k = (int)(r % KH2);
    float v = 0.f;
    if (k < 200) {
      if (n < 200)      v = wq[n * 200 + k];
      else if (n < 400) v = wk[(n - 200) * 200 + k];
      else if (n < 600) v = wv[(n - 400) * 200 + k];
    }
    Wqkv[r] = (_Float16)v;
  } else if (i < S5) {
    int n = (int)(i - S4);
    float v = 0.f;
    if (n < 200) v = bq[n]; else if (n < 400) v = bk[n - 200]; else if (n < 600) v = bv[n - 400];
    bqkv[n] = v;
  } else if (i < S6) {
    long long r = i - S5; int n = (int)(r / KATT), k = (int)(r % KATT);
    dw1t[r] = (_Float16)((k < 10000) ? dw1[(long long)n * 10000 + k] : 0.f);
  }
}

// ---------------- WMMA f16 GEMM: C(MxN) = A(MxK) * B(NxK)^T + bias ------------------
// Block = 8 waves = 128 M-rows x one N strip of NT 16-col tiles (NT compile-time).
// B strip staged through LDS via async global->LDS (ASYNCcnt); A streams from global.
// Requires: M % 128 == 0, K % 32 == 0.
template<int NT, bool F16OUT, bool RELU>
__global__ void gemm_wmma(const _Float16* __restrict__ A, const _Float16* __restrict__ Bw,
                          float* __restrict__ Cf, _Float16* __restrict__ Ch,
                          const float* __restrict__ bias,
                          int N, int K, int lda, int ldb, int ldc,
                          int tn_base, int nstrips) {
  __shared__ __attribute__((aligned(16))) _Float16 bsh[64 * BROW];  // 5 KB
  int tid  = threadIdx.x;
  int wid  = tid >> 5;
  int lane = tid & 31;
  int g    = lane >> 4;
  int lr   = lane & 15;
  int sn   = blockIdx.x % nstrips;
  int mb   = blockIdx.x / nstrips;
  int tn0  = tn_base + (sn << 6);
  int tm   = (mb << 7) + (wid << 4);
  const _Float16* arow = A + (size_t)(tm + lr) * lda;
  // cooperative async B-strip copy: thread -> one b128 (8 halves) per K-chunk
  int ccol  = tid >> 2;            // 0..63
  int cpart = (tid & 3) << 3;      // 0,8,16,24 halves
  int crow  = tn0 + ccol; if (crow > N - 1) crow = N - 1;  // clamp (LDS garbage unused)
  const _Float16* gsrc = Bw + (size_t)crow * ldb + cpart;
  unsigned ldsdst = (unsigned)(uintptr_t)(bsh + ccol * BROW + cpart);

  v8f acc[NT] = {};
  for (int k0 = 0; k0 < K; k0 += 32) {
    __syncthreads();   // protect LDS strip from previous iteration's readers
    asm volatile("global_load_async_to_lds_b128 %0, %1, off"
                 :: "v"(ldsdst), "v"(gsrc + k0) : "memory");
    asm volatile("s_wait_asynccnt 0x0" ::: "memory");
    __syncthreads();
    Frag fa;
    fa.h[0] = *(const v8h*)(arow + k0 + g * 8);
    fa.h[1] = *(const v8h*)(arow + k0 + 16 + g * 8);
    #pragma unroll
    for (int j = 0; j < NT; j++) {
      const _Float16* bl = bsh + (j * 16 + lr) * BROW + g * 8;
      Frag fb;
      fb.h[0] = *(const v8h*)bl;
      fb.h[1] = *(const v8h*)(bl + 16);
      acc[j] = __builtin_amdgcn_wmma_f32_16x16x32_f16(false, fa.v, false, fb.v,
                                                      (short)0, acc[j], false, false);
    }
  }
  #pragma unroll
  for (int j = 0; j < NT; j++) {
    int col = tn0 + j * 16 + lr;
    float bv = bias ? bias[col] : 0.f;
    #pragma unroll
    for (int r = 0; r < 8; r++) {
      int row = tm + g * 8 + r;
      float v = acc[j][r] + bv;
      if (RELU) v = fmaxf(v, 0.f);
      if (F16OUT) Ch[(size_t)row * ldc + col] = (_Float16)v;
      else        Cf[(size_t)row * ldc + col] = v;
    }
  }
}

// host-side dispatch: main strips (NT=4) + optional 32-col tail (NT=2)
static inline void launch_gemm(const _Float16* A, const _Float16* Bw, float* Cf, _Float16* Ch,
                               const float* bias, int M, int N, int K,
                               int lda, int ldb, int ldc, bool relu, hipStream_t stream) {
  int mblocks = M >> 7;
  int full    = N >> 6;
  int tail    = (N & 63) >> 4;   // 0 or 2 for our shapes
  if (full > 0) {
    int blocks = mblocks * full;
    if (Ch)        gemm_wmma<4, true,  false><<<blocks, 256, 0, stream>>>(A, Bw, nullptr, Ch, bias, N, K, lda, ldb, ldc, 0, full);
    else if (relu) gemm_wmma<4, false, true ><<<blocks, 256, 0, stream>>>(A, Bw, Cf, nullptr, bias, N, K, lda, ldb, ldc, 0, full);
    else           gemm_wmma<4, false, false><<<blocks, 256, 0, stream>>>(A, Bw, Cf, nullptr, bias, N, K, lda, ldb, ldc, 0, full);
  }
  if (tail == 2) {
    int tb = full << 6;
    if (Ch)        gemm_wmma<2, true,  false><<<mblocks, 256, 0, stream>>>(A, Bw, nullptr, Ch, bias, N, K, lda, ldb, ldc, tb, 1);
    else if (relu) gemm_wmma<2, false, true ><<<mblocks, 256, 0, stream>>>(A, Bw, Cf, nullptr, bias, N, K, lda, ldb, ldc, tb, 1);
    else           gemm_wmma<2, false, false><<<mblocks, 256, 0, stream>>>(A, Bw, Cf, nullptr, bias, N, K, lda, ldb, ldc, tb, 1);
  }
}

// ---------------- LSTM step-GEMM strip: 1 m-tile x NT n-tiles (compile-time NT) ------
template<int NT>
__device__ __forceinline__ void lstm_strip(const _Float16* hsh, const _Float16* Whh,
                                           float* gsh, const _Float16* gxp,
                                           int tm, int tn0, int lr, int g) {
  const _Float16* arow = hsh + (tm + lr) * KHID;
  v8f acc[NT] = {};
  #pragma unroll
  for (int k0 = 0; k0 < KHID; k0 += 32) {
    Frag fa;
    fa.h[0] = *(const v8h*)(arow + k0 + g * 8);
    fa.h[1] = *(const v8h*)(arow + k0 + 16 + g * 8);
    #pragma unroll
    for (int j = 0; j < NT; j++) {
      const _Float16* bl = Whh + (size_t)(tn0 + j * 16 + lr) * KHID + k0 + g * 8;
      Frag fb;
      fb.h[0] = *(const v8h*)bl;
      fb.h[1] = *(const v8h*)(bl + 16);
      acc[j] = __builtin_amdgcn_wmma_f32_16x16x32_f16(false, fa.v, false, fb.v,
                                                      (short)0, acc[j], false, false);
    }
  }
  #pragma unroll
  for (int j = 0; j < NT; j++) {
    int col = tn0 + j * 16 + lr;
    #pragma unroll
    for (int r = 0; r < 8; r++) {
      int m = tm + g * 8 + r;
      float gxv = (float)gxp[(size_t)m * L_ * NGATE + col];
      gsh[m * 400 + col] = acc[j][r] + gxv;
    }
  }
}

// ---------------- BiLSTM: block = 32 batch rows, one direction; 50 sequential steps --
// h in LDS (f16, K padded to 128), c in registers, gates staged in LDS f32.
__global__ void lstm_kernel(const _Float16* __restrict__ gx16,
                            const _Float16* __restrict__ WhhF,
                            const _Float16* __restrict__ WhhB,
                            _Float16* __restrict__ hcat) {
  __shared__ __attribute__((aligned(16))) _Float16 hsh[32 * KHID];  // 8 KB
  __shared__ float gsh[32 * 400];                                   // 50 KB
  int tid  = threadIdx.x;
  int wid  = tid >> 5;
  int lane = tid & 31;
  int g    = lane >> 4;
  int lr   = lane & 15;
  int dir  = blockIdx.y;
  int b0   = blockIdx.x * 32;
  const _Float16* Whh = dir ? WhhB : WhhF;
  for (int i = tid; i < 32 * KHID; i += 256) hsh[i] = (_Float16)0.f;
  float creg[13];
  #pragma unroll
  for (int i = 0; i < 13; i++) creg[i] = 0.f;
  __syncthreads();

  for (int s = 0; s < L_; s++) {
    int tt = dir ? (L_ - 1 - s) : s;
    const _Float16* gxp = gx16 + ((size_t)b0 * L_ + tt) * NGATE + dir * 400;
    // gates = h @ Whh^T + gx : 12 NT=4 strips (cols 0..383) + 2 NT=1 strips (cols 384..399)
    for (int strip = wid; strip < 12; strip += 8)
      lstm_strip<4>(hsh, Whh, gsh, gxp, (strip / 6) << 4, (strip % 6) << 6, lr, g);
    if (wid >= 6)
      lstm_strip<1>(hsh, Whh, gsh, gxp, (wid - 6) << 4, 384, lr, g);
    __syncthreads();
    // activations + state update (torch gate order i,f,g,o)
    int ri = 0;
    for (int idx = tid; idx < 32 * H_; idx += 256, ri++) {
      int m = idx / H_;
      int j = idx - m * H_;
      float ig = gsh[m * 400 + j];
      float fg = gsh[m * 400 + 100 + j];
      float gg = gsh[m * 400 + 200 + j];
      float og = gsh[m * 400 + 300 + j];
      float c = sigf(fg) * creg[ri] + sigf(ig) * tanhfast(gg);
      creg[ri] = c;
      float h = sigf(og) * tanhfast(c);
      hsh[m * KHID + j] = (_Float16)h;
      hcat[((size_t)(b0 + m) * L_ + tt) * KH2 + dir * H_ + j] = (_Float16)h;
    }
    __syncthreads();
  }
}

// ---------------- AFT-full attention (exp factorized) --------------------------------
__global__ void aft_kernel(const float* __restrict__ qkv, const float* __restrict__ pos_bias,
                           _Float16* __restrict__ att16) {
  int idx = blockIdx.x * blockDim.x + threadIdx.x;
  int total = B_ * L_ * D_;
  if (idx >= total) return;
  int d = idx % D_;
  int i = (idx / D_) % L_;
  int b = idx / (D_ * L_);
  const float* base = qkv + (size_t)b * L_ * NQKV;
  float q = base[(size_t)i * NQKV + d];
  float num = 0.f, den = 0.f;
  for (int j = 0; j < L_; j++) {
    float kk = base[(size_t)j * NQKV + 200 + d];
    float vv = base[(size_t)j * NQKV + 400 + d];
    float w  = __expf(pos_bias[i * L_ + j] + kk);
    num += w * vv;
    den += w;
  }
  float att = sigf(q) * (num / den);
  att16[(size_t)b * KATT + i * D_ + d] = (_Float16)att;
}

// ---------------- final tiny dense (128 -> 5) + sigmoid ------------------------------
__global__ void dense2_kernel(const float* __restrict__ hid, const float* __restrict__ dw2,
                              const float* __restrict__ db2, float* __restrict__ out) {
  int idx = blockIdx.x * blockDim.x + threadIdx.x;
  if (idx >= B_ * 5) return;
  int o = idx % 5, b = idx / 5;
  float acc = db2[o];
  const float* hrow = hid + (size_t)b * 128;
  const float* wrow = dw2 + (size_t)o * 128;
  for (int k = 0; k < 128; k++) acc += hrow[k] * wrow[k];
  out[idx] = sigf(acc);
}

extern "C" void kernel_launch(void* const* d_in, const int* in_sizes, int n_in,
                              void* d_out, int out_size, void* d_ws, size_t ws_size,
                              hipStream_t stream) {
  (void)in_sizes; (void)n_in; (void)out_size; (void)ws_size;
  const int*   x     = (const int*)d_in[0];
  const float* emb1  = (const float*)d_in[1];
  const float* emb2  = (const float*)d_in[2];
  const float* cw2   = (const float*)d_in[3];
  const float* cb2   = (const float*)d_in[4];
  const float* cw3   = (const float*)d_in[5];
  const float* cb3   = (const float*)d_in[6];
  const float* cw8   = (const float*)d_in[7];
  const float* cb8   = (const float*)d_in[8];
  const float* wih_f = (const float*)d_in[9];
  const float* whh_f = (const float*)d_in[10];
  const float* bih_f = (const float*)d_in[11];
  const float* bhh_f = (const float*)d_in[12];
  const float* wih_b = (const float*)d_in[13];
  const float* whh_b = (const float*)d_in[14];
  const float* bih_b = (const float*)d_in[15];
  const float* bhh_b = (const float*)d_in[16];
  const float* wq    = (const float*)d_in[17];
  const float* bq    = (const float*)d_in[18];
  const float* wk    = (const float*)d_in[19];
  const float* bk    = (const float*)d_in[20];
  const float* wv    = (const float*)d_in[21];
  const float* bv    = (const float*)d_in[22];
  const float* pos_bias = (const float*)d_in[23];
  const float* dw1   = (const float*)d_in[24];
  const float* db1   = (const float*)d_in[25];
  const float* dw2   = (const float*)d_in[26];
  const float* db2   = (const float*)d_in[27];
  float* out = (float*)d_out;

  char* ws = (char*)d_ws;
  size_t off = 0;
  auto carve = [&](size_t bytes) -> char* {
    char* p = ws + off;
    off = (off + bytes + 255) & ~(size_t)255;
    return p;
  };
  const size_t MROWS = (size_t)B_ * L_;  // 51200
  _Float16* seq16  = (_Float16*)carve(MROWS * KSEQ * 2);
  _Float16* WihT   = (_Float16*)carve((size_t)NGATE * KSEQ * 2);
  float*    gxbias = (float*)   carve((size_t)NGATE * 4);
  _Float16* WhhF   = (_Float16*)carve((size_t)400 * KHID * 2);
  _Float16* WhhB   = (_Float16*)carve((size_t)400 * KHID * 2);
  _Float16* Wqkv   = (_Float16*)carve((size_t)NQKV * KH2 * 2);
  float*    bqkv   = (float*)   carve((size_t)NQKV * 4);
  _Float16* dw1t   = (_Float16*)carve((size_t)128 * KATT * 2);
  float*    conv2r = (float*)   carve((size_t)B_ * CD_ * 49 * 4);
  float*    conv3r = (float*)   carve((size_t)B_ * CD_ * 48 * 4);
  float*    conv8r = (float*)   carve((size_t)B_ * CD_ * 43 * 4);
  _Float16* gx16   = (_Float16*)carve(MROWS * NGATE * 2);
  _Float16* hcat16 = (_Float16*)carve(MROWS * KH2 * 2);
  float*    qkvbuf = (float*)   carve(MROWS * NQKV * 4);
  _Float16* att16  = (_Float16*)carve((size_t)B_ * KATT * 2);
  float*    hidbuf = (float*)   carve((size_t)B_ * 128 * 4);

  // zero buffers whose padding is never written
  hipMemsetAsync(seq16,  0, MROWS * KSEQ * 2, stream);
  hipMemsetAsync(hcat16, 0, MROWS * KH2  * 2, stream);
  hipMemsetAsync(att16,  0, (size_t)B_ * KATT * 2, stream);

  // weight prep
  {
    long long total = 800LL * KSEQ + 800 + 2 * 400LL * KHID +
                      (long long)NQKV * KH2 + NQKV + 128LL * KATT;
    int blocks = (int)((total + 255) / 256);
    prep_kernel<<<blocks, 256, 0, stream>>>(wih_f, wih_b, bih_f, bhh_f, bih_b, bhh_b,
                                            whh_f, whh_b, wq, wk, wv, bq, bk, bv, dw1,
                                            WihT, gxbias, WhhF, WhhB, Wqkv, bqkv, dw1t);
  }
  // embeddings
  embed_kernel<<<(B_ * L_ + 255) / 256, 256, 0, stream>>>(x, emb1, emb2, seq16);
  // conv branches (raw conv+relu), then maxpool+pad into seq16 channels
  conv_kernel<<<(B_ * CD_ * 49 + 255) / 256, 256, 0, stream>>>(seq16, cw2, cb2, conv2r, 2, 49);
  conv_kernel<<<(B_ * CD_ * 48 + 255) / 256, 256, 0, stream>>>(seq16, cw3, cb3, conv3r, 3, 48);
  conv_kernel<<<(B_ * CD_ * 43 + 255) / 256, 256, 0, stream>>>(seq16, cw8, cb8, conv8r, 8, 43);
  pool_kernel<<<(B_ * CD_ * 47 + 255) / 256, 256, 0, stream>>>(conv2r, seq16, 49, 63);
  pool_kernel<<<(B_ * CD_ * 46 + 255) / 256, 256, 0, stream>>>(conv3r, seq16, 48, 123);
  pool_kernel<<<(B_ * CD_ * 41 + 255) / 256, 256, 0, stream>>>(conv8r, seq16, 43, 183);
  // gx = seq @ [wih_f; wih_b]^T + bias   (f16 out)
  launch_gemm(seq16, WihT, nullptr, gx16, gxbias, (int)MROWS, NGATE, KSEQ,
              KSEQ, KSEQ, NGATE, false, stream);
  // bidirectional LSTM
  lstm_kernel<<<dim3(B_ / 32, 2), 256, 0, stream>>>(gx16, WhhF, WhhB, hcat16);
  // fused QKV GEMM
  launch_gemm(hcat16, Wqkv, qkvbuf, nullptr, bqkv, (int)MROWS, NQKV, KH2,
              KH2, KH2, NQKV, false, stream);
  // AFT attention
  aft_kernel<<<(B_ * L_ * D_ + 255) / 256, 256, 0, stream>>>(qkvbuf, pos_bias, att16);
  // dense1 (relu)
  launch_gemm(att16, dw1t, hidbuf, nullptr, db1, B_, 128, KATT,
              KATT, KATT, 128, true, stream);
  // dense2 + sigmoid -> output (B,5) f32
  dense2_kernel<<<(B_ * 5 + 255) / 256, 256, 0, stream>>>(hidbuf, dw2, db2, out);
}
